// Streamed_30004641530482
// MI455X (gfx1250) — compile-verified, matrix operand order checked by reading, not run on hardware
//
#include <hip/hip_runtime.h>
#include <hip/hip_bf16.h>
#include <stdint.h>

// CDNA5 WMMA operand types
typedef __attribute__((ext_vector_type(16))) __bf16 v16bf;
typedef __attribute__((ext_vector_type(8)))  float  v8f;
typedef __attribute__((ext_vector_type(4)))  int    v4i;

typedef __attribute__((address_space(1))) v4i glb_v4i;  // global (prints as __device__)
typedef __attribute__((address_space(3))) v4i lds_v4i;  // LDS (prints as __shared__)

union BFV {
    v16bf    v;
    uint32_t d[8];
    uint4    q[2];
};

__device__ __forceinline__ float bft(float a) {
    // truncate f32 -> bf16 (kept as f32 bits)
    return __uint_as_float(__float_as_uint(a) & 0xFFFF0000u);
}
__device__ __forceinline__ uint32_t pkbf(float a, float b) {
    // dword = {b[31:16], a[31:16]} in ONE v_perm_b32
    return __builtin_amdgcn_perm(__float_as_uint(b), __float_as_uint(a), 0x07060302u);
}
// Split 8 consecutive floats into 4 packed-hi dwords + 4 packed-lo dwords
__device__ __forceinline__ void split8(const float* __restrict__ p,
                                       uint32_t* hi, uint32_t* lo) {
    float4 f0 = *(const float4*)(p);
    float4 f1 = *(const float4*)(p + 4);
    float v[8] = {f0.x, f0.y, f0.z, f0.w, f1.x, f1.y, f1.z, f1.w};
#pragma unroll
    for (int i = 0; i < 4; ++i) {
        float a = v[2 * i], b = v[2 * i + 1];
        hi[i] = pkbf(a, b);
        lo[i] = pkbf(a - bft(a), b - bft(b));
    }
}

#define WMMA_BF16(A, Bm, C) \
    __builtin_amdgcn_wmma_f32_16x16x32_bf16(false, (A), false, (Bm), (short)0, (C), false, false)

#if __has_builtin(__builtin_amdgcn_global_load_async_to_lds_b128)
#define HAVE_ASYNC_LDS 1
#else
#define HAVE_ASYNC_LDS 0
#endif

__device__ __forceinline__ void async_wait0() {
#if __has_builtin(__builtin_amdgcn_s_wait_asynccnt)
    __builtin_amdgcn_s_wait_asynccnt(0);
#else
    asm volatile("s_wait_asynccnt 0" ::: "memory");
#endif
}

// Copy one 64-token x slab (64*128 f32 = 32KB, contiguous) into LDS staging.
__device__ __forceinline__ void prefetch_batch(const float* __restrict__ gsrc,
                                               float* __restrict__ lraw, int tid) {
#pragma unroll
    for (int k = 0; k < 8; ++k) {
        const int off = (tid + k * 256) * 4;  // 16B chunks, 2048 total
#if HAVE_ASYNC_LDS
        __builtin_amdgcn_global_load_async_to_lds_b128(
            (glb_v4i*)(gsrc + off),
            (lds_v4i*)(lraw + off), 0, 0);
#else
        *(float4*)(lraw + off) = *(const float4*)(gsrc + off);
#endif
    }
}

// Fused 2-layer expert MLP: out = relu(W2 @ relu(W1 @ x^T + b1) + b2)^T
// block = 256 threads (8 wave32), handles 512 tokens of one expert.
__global__ __launch_bounds__(256, 1)
void moe_mlp2_bf16x3(const float* __restrict__ x,
                     const float* __restrict__ W,
                     const float* __restrict__ Bv,
                     float* __restrict__ out) {
    // x staged as B-operands (hi/lo bf16): [tile][kstep][part][lane][8 dwords]
    __shared__ __align__(16) uint32_t xbuf[4][4][2][32][8];  // 32 KB
    // layer-1 activations, already in layer-2 B-operand layout
    __shared__ __align__(16) uint32_t zbuf[4][4][2][32][8];  // 32 KB
    extern __shared__ float xraw[];                          // 32 KB dynamic: raw f32 x slab

    const int tid  = threadIdx.x;
    const int wave = tid >> 5;
    const int lane = tid & 31;
    const int ln   = lane & 15;   // token / M row within tile
    const int kh   = lane >> 4;   // K-half select (WMMA operand layout)

    const int  model = blockIdx.x >> 6;        // 64 blocks per expert
    const int  blk   = blockIdx.x & 63;
    const long tok0  = (long)model * 32768 + (long)blk * 512;

    // ---- persistent per-wave weights: A-operand tiles (rows 16*wave..+15), hi/lo split
    BFV a1hi[4], a1lo[4], a2hi[4], a2lo[4];
    const float* W1row = W + ((size_t)(model * 2 + 0) * 128 + 16 * wave + ln) * 128;
    const float* W2row = W + ((size_t)(model * 2 + 1) * 128 + 16 * wave + ln) * 128;
#pragma unroll
    for (int j = 0; j < 4; ++j) {
        const int c0 = 32 * j + 8 * kh;
        split8(W1row + c0,      a1hi[j].d,     a1lo[j].d);
        split8(W1row + c0 + 16, a1hi[j].d + 4, a1lo[j].d + 4);
        split8(W2row + c0,      a2hi[j].d,     a2lo[j].d);
        split8(W2row + c0 + 16, a2hi[j].d + 4, a2lo[j].d + 4);
    }
    float bias1[8], bias2[8];
    {
        const float* b1 = Bv + (size_t)(model * 2 + 0) * 128 + 16 * wave + 8 * kh;
        const float* b2 = Bv + (size_t)(model * 2 + 1) * 128 + 16 * wave + 8 * kh;
#pragma unroll
        for (int v = 0; v < 8; ++v) { bias1[v] = b1[v]; bias2[v] = b2[v]; }
    }

    const int jw = wave >> 1;  // which layer-2 K-step this wave's z-tile feeds
    const int hf = wave & 1;   // low/high element half of that K-step

    // kick off async prefetch of batch 0
    prefetch_batch(x + tok0 * 128, xraw, tid);

    for (int nb = 0; nb < 8; ++nb) {
        const long btok = tok0 + nb * 64;

#if HAVE_ASYNC_LDS
        async_wait0();            // this wave's async chunks have landed in LDS
#endif
        __syncthreads();          // everyone's chunks have landed -> xraw valid

        // ---- Phase A: convert xraw -> hi/lo bf16 B-operands (16 units over 8 waves)
#pragma unroll
        for (int uu = 0; uu < 2; ++uu) {
            const int u  = wave * 2 + uu;
            const int ct = u >> 2;        // tile
            const int cj = u & 3;         // K-step
            const float* xr = xraw + (ct * 16 + ln) * 128;
            const int c0 = 32 * cj + 8 * kh;
            uint32_t hi[8], lo[8];
            split8(xr + c0,      hi,     lo);
            split8(xr + c0 + 16, hi + 4, lo + 4);
            *(uint4*)&xbuf[ct][cj][0][lane][0] = *(uint4*)&hi[0];
            *(uint4*)&xbuf[ct][cj][0][lane][4] = *(uint4*)&hi[4];
            *(uint4*)&xbuf[ct][cj][1][lane][0] = *(uint4*)&lo[0];
            *(uint4*)&xbuf[ct][cj][1][lane][4] = *(uint4*)&lo[4];
        }
        __syncthreads();          // xbuf ready; xraw free for the next prefetch

        if (nb < 7) prefetch_batch(x + (btok + 64) * 128, xraw, tid);

        // ---- Phase B: layer 1.  z^T = relu(W1 @ x^T + b1), packed lane-locally
        for (int tt = 0; tt < 4; ++tt) {
            v8f acc = {0.f, 0.f, 0.f, 0.f, 0.f, 0.f, 0.f, 0.f};
#pragma unroll
            for (int j = 0; j < 4; ++j) {
                BFV bh, bl;
                bh.q[0] = *(const uint4*)&xbuf[tt][j][0][lane][0];
                bh.q[1] = *(const uint4*)&xbuf[tt][j][0][lane][4];
                bl.q[0] = *(const uint4*)&xbuf[tt][j][1][lane][0];
                bl.q[1] = *(const uint4*)&xbuf[tt][j][1][lane][4];
                acc = WMMA_BF16(a1hi[j].v, bh.v, acc);
                acc = WMMA_BF16(a1hi[j].v, bl.v, acc);
                acc = WMMA_BF16(a1lo[j].v, bh.v, acc);
            }
            uint32_t zh[4], zl[4];
#pragma unroll
            for (int d2 = 0; d2 < 4; ++d2) {
                float a = acc[2 * d2]     + bias1[2 * d2];
                float b = acc[2 * d2 + 1] + bias1[2 * d2 + 1];
                a = a > 0.f ? a : 0.f;
                b = b > 0.f ? b : 0.f;
                zh[d2] = pkbf(a, b);
                zl[d2] = pkbf(a - bft(a), b - bft(b));
            }
            *(uint4*)&zbuf[tt][jw][0][lane][hf * 4] = *(uint4*)&zh[0];
            *(uint4*)&zbuf[tt][jw][1][lane][hf * 4] = *(uint4*)&zl[0];
        }
        __syncthreads();

        // ---- Phase C: layer 2 + fp32 store
        for (int tt = 0; tt < 4; ++tt) {
            v8f acc = {0.f, 0.f, 0.f, 0.f, 0.f, 0.f, 0.f, 0.f};
#pragma unroll
            for (int j = 0; j < 4; ++j) {
                BFV bh, bl;
                bh.q[0] = *(const uint4*)&zbuf[tt][j][0][lane][0];
                bh.q[1] = *(const uint4*)&zbuf[tt][j][0][lane][4];
                bl.q[0] = *(const uint4*)&zbuf[tt][j][1][lane][0];
                bl.q[1] = *(const uint4*)&zbuf[tt][j][1][lane][4];
                acc = WMMA_BF16(a2hi[j].v, bh.v, acc);
                acc = WMMA_BF16(a2hi[j].v, bl.v, acc);
                acc = WMMA_BF16(a2lo[j].v, bh.v, acc);
            }
            float r[8];
#pragma unroll
            for (int v = 0; v < 8; ++v) {
                float a = acc[v] + bias2[v];
                r[v] = a > 0.f ? a : 0.f;
            }
            float* orow = out + (btok + tt * 16 + ln) * 128 + 16 * wave + 8 * kh;
            *(float4*)(orow)     = make_float4(r[0], r[1], r[2], r[3]);
            *(float4*)(orow + 4) = make_float4(r[4], r[5], r[6], r[7]);
        }
        __syncthreads();
    }
}

extern "C" void kernel_launch(void* const* d_in, const int* in_sizes, int n_in,
                              void* d_out, int out_size, void* d_ws, size_t ws_size,
                              hipStream_t stream) {
    const float* x = (const float*)d_in[0];   // [524288,128] f32
    const float* W = (const float*)d_in[1];   // [16,2,128,128] f32
    const float* b = (const float*)d_in[2];   // [16,2,128] f32
    float* out = (float*)d_out;               // [524288,128] f32

    dim3 grid(16 * 64);   // 16 experts x 64 blocks (512 tokens each)
    dim3 block(256);      // 8 wave32
    hipLaunchKernelGGL(moe_mlp2_bf16x3, grid, block, 32768 /* xraw */, stream,
                       x, W, b, out);
}
